// MultiHeadAttention_80564996538581
// MI455X (gfx1250) — matrix operand order repaired and checked by reading, not compiled, not run on hardware
//
#include <hip/hip_runtime.h>
#include <hip/hip_bf16.h>

typedef __attribute__((ext_vector_type(16))) __bf16 v16bf;
typedef __attribute__((ext_vector_type(8)))  __bf16 v8bf;
typedef __attribute__((ext_vector_type(8)))  float  v8f;
typedef __attribute__((ext_vector_type(4)))  float  v4f;
typedef __attribute__((ext_vector_type(4)))  int    v4i;

#define D_MODEL 512
#define N_HEADS 8
#define D_HEAD  64
#define SEQ     2048
#define BATCH   4
#define M_TOTAL (BATCH*SEQ)

__device__ __forceinline__ v8f wmma_bf16(v16bf a, v16bf b, v8f c) {
    return __builtin_amdgcn_wmma_f32_16x16x32_bf16(
        /*neg_a=*/false, a, /*neg_b=*/false, b,
        /*c_mod=*/(short)0, c, /*reuse_a=*/false, /*reuse_b=*/false);
}

// ---------------------------------------------------------------------------
// CDNA5 async global->LDS copy (ASYNCcnt-tracked DMA), guarded for
// compile-safety. Builtin prototype (from compiler diagnostic): the pointers
// are v4i* in global (AS1) / LDS (AS3) address spaces.
// ---------------------------------------------------------------------------
#if defined(__HIP_DEVICE_COMPILE__) && __has_builtin(__builtin_amdgcn_global_load_async_to_lds_b128)
#define HAVE_ASYNC_LDS 1
typedef __attribute__((address_space(1))) v4i* gv4i_p;
typedef __attribute__((address_space(3))) v4i* sv4i_p;
#else
#define HAVE_ASYNC_LDS 0
#endif

__device__ __forceinline__ void async_cp16(void* lds, const void* g) {
#if HAVE_ASYNC_LDS
    __builtin_amdgcn_global_load_async_to_lds_b128(
        (gv4i_p)(void*)g, (sv4i_p)lds, /*offset=*/0, /*cpol=*/0);
#else
    *(v8bf*)lds = *(const v8bf*)g;
#endif
}

template <int N>
__device__ __forceinline__ void wait_async_le() {
#if HAVE_ASYNC_LDS
#if __has_builtin(__builtin_amdgcn_s_wait_asynccnt)
    __builtin_amdgcn_s_wait_asynccnt(N);
#else
    asm volatile("s_wait_asynccnt %0" ::"i"(N) : "memory");
#endif
#endif
}

// ---------------------------------------------------------------------------
// Pack W (row-major [512,512] f32) into bf16 B-fragment order for X @ W^T:
// B[k][n] = W[n][k].  Fragment f = nt*16 + kt. Per lane: k 0..7 | 16..23 of the 32 (lanes>=16: +8).
// ---------------------------------------------------------------------------
__global__ void pack_weights(const float* __restrict__ Wq, const float* __restrict__ Wk,
                             const float* __restrict__ Wv, const float* __restrict__ Wo,
                             __bf16* __restrict__ wpack) {
    const int f    = blockIdx.x;           // 0..511
    const int z    = blockIdx.y;           // 0..3
    const float* W = (z == 0) ? Wq : (z == 1) ? Wk : (z == 2) ? Wv : Wo;
    const int lane = threadIdx.x;
    const int nt = f >> 4, kt = f & 15;
    const int n  = nt * 16 + (lane & 15);
    const int k0 = kt * 32 + ((lane < 16) ? 0 : 8);
    __bf16* dst      = wpack + ((size_t)(z * 512 + f) * 32 + lane) * 16;
    const float* src = W + (size_t)n * D_MODEL + k0;
#pragma unroll
    for (int e = 0; e < 16; ++e) dst[e] = (__bf16)src[(e < 8) ? e : e + 8];
}

// ---------------------------------------------------------------------------
// GEMM: out[m,n] = sum_k A[m,k]*W[n,k] + bias[n].  One wave = 16x64 tile.
// mode 0: bf16 out, head-split [B,H,S,64]
// mode 1: bf16 out, head-split transposed [B,H,64,S]   (for V)
// mode 2: f32 out, row-major [M,512]
// ---------------------------------------------------------------------------
template <typename AT>
__global__ void __launch_bounds__(128)
gemm_wmma(const AT* __restrict__ A, const __bf16* __restrict__ wpk,
          const float* __restrict__ bias, void* __restrict__ out, int mode) {
    const int lane = threadIdx.x & 31;
    const int warp = threadIdx.x >> 5;
    const int m0   = blockIdx.x * 64 + warp * 16;
    const int n0   = blockIdx.y * 64;
    const int lo   = lane & 15;
    const int hi8  = (lane < 16) ? 0 : 8;

    v8f acc[4] = {};

#pragma unroll 4
    for (int kt = 0; kt < D_MODEL / 32; ++kt) {
        // A fragment 16x32: lane<16 -> K 0..7 | 16..23 ; lane>=16 -> +8
        v16bf a;
        const int kb = kt * 32 + ((lane < 16) ? 0 : 8);
        if constexpr (sizeof(AT) == 4) {
            const float* ar = (const float*)A + (size_t)(m0 + lo) * D_MODEL + kb;
            const v4f f0 = *(const v4f*)(ar);
            const v4f f1 = *(const v4f*)(ar + 4);
            const v4f f2 = *(const v4f*)(ar + 16);
            const v4f f3 = *(const v4f*)(ar + 20);
#pragma unroll
            for (int e = 0; e < 4; ++e) {
                a[e]      = (__bf16)f0[e];
                a[e + 4]  = (__bf16)f1[e];
                a[e + 8]  = (__bf16)f2[e];
                a[e + 12] = (__bf16)f3[e];
            }
        } else {
            const __bf16* ar = (const __bf16*)A + (size_t)(m0 + lo) * D_MODEL + kb;
            v8bf c0 = *(const v8bf*)(ar);
            v8bf c1 = *(const v8bf*)(ar + 16);
#pragma unroll
            for (int e = 0; e < 8; ++e) { a[e] = c0[e]; a[e + 8] = c1[e]; }
        }
#pragma unroll
        for (int nt = 0; nt < 4; ++nt) {
            const int f = (n0 / 16 + nt) * 16 + kt;
            const v16bf b = *(const v16bf*)(wpk + ((size_t)f * 32 + lane) * 16);
            acc[nt] = wmma_bf16(a, b, acc[nt]);
        }
    }

#pragma unroll
    for (int nt = 0; nt < 4; ++nt) {
#pragma unroll
        for (int r = 0; r < 8; ++r) {
            const int m = m0 + r + hi8;
            const int n = n0 + nt * 16 + lo;
            const float val = acc[nt][r] + bias[n];
            if (mode == 2) {
                ((float*)out)[(size_t)m * D_MODEL + n] = val;
            } else {
                const int b = m >> 11, s = m & (SEQ - 1);
                const int h = n >> 6, dh = n & (D_HEAD - 1);
                __bf16* ob = (__bf16*)out;
                if (mode == 0)
                    ob[((size_t)(b * N_HEADS + h) * SEQ + s) * D_HEAD + dh] = (__bf16)val;
                else
                    ob[((size_t)(b * N_HEADS + h) * D_HEAD + dh) * SEQ + s] = (__bf16)val;
            }
        }
    }
}

// ---------------------------------------------------------------------------
// Flash attention. Block = 4 waves = 4 consecutive 16-row q-tiles of the SAME
// (b,h): the block shares one K/V stream. Each 32-key tile (K: 32x64, V(T):
// 64x32, 8 KiB total) is staged into LDS once per block via async DMA,
// double-buffered; each wave issues a quarter (4x b128) and ASYNCcnt waits +
// two barriers hand the buffer over.
// q/k: [B,H,S,64] bf16, vT: [B,H,64,S] bf16, out: [B,S,512] bf16.
// ---------------------------------------------------------------------------
__global__ void __launch_bounds__(128)
flash_attn(const __bf16* __restrict__ qbf, const __bf16* __restrict__ kbf,
           const __bf16* __restrict__ vtbf, const int* __restrict__ mask,
           __bf16* __restrict__ attn_out) {
    __shared__ __attribute__((aligned(32))) __bf16 bufK[2][32 * 64];  // [key][dh]
    __shared__ __attribute__((aligned(32))) __bf16 bufV[2][64 * 32];  // [dh][key]
    __shared__ __attribute__((aligned(32))) __bf16 plds[4][16 * 32];  // per-wave P tile

    const int tid  = threadIdx.x;
    const int lane = tid & 31;
    const int warp = tid >> 5;
    const int wq   = blockIdx.x * 4 + warp;  // 0..4095
    const int bh   = wq >> 7;                // block-uniform (0..31)
    const int qt   = wq & 127;
    const int b = bh >> 3, h = bh & 7;
    const int q0  = qt * 16;
    const int lo  = lane & 15;
    const int hi8 = (lane < 16) ? 0 : 8;

    const __bf16* kbase = kbf + (size_t)bh * SEQ * D_HEAD;    // block-uniform base
    const __bf16* vbase = vtbf + (size_t)bh * D_HEAD * SEQ;   // block-uniform base

    // Q fragments: 16x64 -> two 16x32 A-frags (contiguous 16B chunks)
    v16bf qa[2];
    {
        const __bf16* qrow = qbf + ((size_t)bh * SEQ + q0 + lo) * D_HEAD;
#pragma unroll
        for (int d = 0; d < 2; ++d) {
            const __bf16* p = qrow + d * 32 + ((lane < 16) ? 0 : 16);
            v8bf c0 = *(const v8bf*)(p);
            v8bf c1 = *(const v8bf*)(p + 8);
#pragma unroll
            for (int e = 0; e < 8; ++e) { qa[d][e] = c0[e]; qa[d][e + 8] = c1[e]; }
        }
    }

    // mask: SGPR base (block-uniform) + precomputed per-lane byte offsets
    const char* mbase = (const char*)mask + (size_t)b * SEQ * SEQ * sizeof(int);
    int moff[2][8];
#pragma unroll
    for (int t = 0; t < 2; ++t)
#pragma unroll
        for (int r = 0; r < 8; ++r)
            moff[t][r] = ((q0 + r + hi8) * SEQ + t * 16 + lo) * (int)sizeof(int);

    v8f oacc[4] = {};
    float mrow[8], lsum[8];
#pragma unroll
    for (int r = 0; r < 8; ++r) { mrow[r] = -1e30f; lsum[r] = 0.f; }

    // stage one 32-key tile (K 4KiB contiguous + V 64 rows x 64B) into buffer
    auto stage = [&](int jt, int bsel) {
        const char* kg = (const char*)kbase + (size_t)jt * 32 * D_HEAD * 2;
#pragma unroll
        for (int rnd = 0; rnd < 4; ++rnd) {
            const int c = rnd * 128 + tid;  // 512 x 16B chunks per tile
            if (c < 256) {
                async_cp16((char*)&bufK[bsel][0] + c * 16, kg + c * 16);
            } else {
                const int c2 = c - 256;
                const int row = c2 >> 2, part = c2 & 3;
                async_cp16((char*)&bufV[bsel][0] + c2 * 16,
                           (const char*)vbase + ((size_t)row * SEQ + jt * 32 + part * 8) * 2);
            }
        }
    };

    constexpr int NT = SEQ / 32;  // 64 iterations of 32 keys
    stage(0, 0);

    for (int j = 0; j < NT; ++j) {
        const int cur = j & 1;
        __syncthreads();                      // everyone done reading buf[cur^1]
        if (j + 1 < NT) {
            stage(j + 1, cur ^ 1);            // queue next tile (4 async ops/wave)
            wait_async_le<4>();               // my 4 ops of tile j are done
        } else {
            wait_async_le<0>();
        }
        __syncthreads();                      // all quarters of tile j in LDS

        const int s0 = j * 32;
        const __bf16* kl = &bufK[cur][0];
        const __bf16* vl = &bufV[cur][0];

        // scores: S = Q(16x64) @ K_j^T -> two 16x16 tiles, 4 WMMAs (LDS-fed)
        v8f sc[2] = {};
#pragma unroll
        for (int t = 0; t < 2; ++t) {
            const __bf16* krow = kl + (t * 16 + lo) * D_HEAD + ((lane < 16) ? 0 : 16);
#pragma unroll
            for (int d = 0; d < 2; ++d) {
                const v16bf kb = *(const v16bf*)(krow + d * 32);
                sc[t] = wmma_bf16(qa[d], kb, sc[t]);
            }
        }

        // scale + mask (saddr-form loads: uniform base + lane offset + s0*4)
        const int mjb = s0 * (int)sizeof(int);
#pragma unroll
        for (int t = 0; t < 2; ++t)
#pragma unroll
            for (int r = 0; r < 8; ++r) {
                const float v = sc[t][r] * 0.125f;  // 1/sqrt(64)
                const int mv = *(const int*)(mbase + (moff[t][r] + mjb));
                sc[t][r] = (mv == 0) ? -1.0e9f : v;
            }

        // online softmax: per-row stats via xor-shuffle within 16-lane halves
#pragma unroll
        for (int r = 0; r < 8; ++r) {
            float mx = fmaxf(sc[0][r], sc[1][r]);
#pragma unroll
            for (int o = 1; o < 16; o <<= 1) mx = fmaxf(mx, __shfl_xor(mx, o, 32));
            const float mnew = fmaxf(mrow[r], mx);
            const float corr = __expf(mrow[r] - mnew);
            mrow[r] = mnew;
            lsum[r] *= corr;
#pragma unroll
            for (int nt = 0; nt < 4; ++nt) oacc[nt][r] *= corr;
            const float p0 = __expf(sc[0][r] - mnew);
            const float p1 = __expf(sc[1][r] - mnew);
            sc[0][r] = p0; sc[1][r] = p1;
            float rs = p0 + p1;
#pragma unroll
            for (int o = 1; o < 16; o <<= 1) rs += __shfl_xor(rs, o, 32);
            lsum[r] += rs;
        }

        // C-layout -> A-layout for P via per-wave LDS tile (same-wave DS in order)
#pragma unroll
        for (int t = 0; t < 2; ++t)
#pragma unroll
            for (int r = 0; r < 8; ++r)
                plds[warp][(r + hi8) * 32 + t * 16 + lo] = (__bf16)sc[t][r];

        v16bf pa;
        {
            const __bf16* pr = &plds[warp][lo * 32 + ((lane < 16) ? 0 : 16)];
            v8bf c0 = *(const v8bf*)(pr);
            v8bf c1 = *(const v8bf*)(pr + 8);
#pragma unroll
            for (int e = 0; e < 8; ++e) { pa[e] = c0[e]; pa[e + 8] = c1[e]; }
        }

        // O += P(16x32) @ V_j(32x64): B-frags contiguous in LDS V tile
#pragma unroll
        for (int nt = 0; nt < 4; ++nt) {
            const v16bf vb = *(const v16bf*)(vl + (nt * 16 + lo) * 32 + ((lane < 16) ? 0 : 16));
            oacc[nt] = wmma_bf16(pa, vb, oacc[nt]);
        }
    }

    // normalize and write concat layout [B,S,512]
#pragma unroll
    for (int nt = 0; nt < 4; ++nt)
#pragma unroll
        for (int r = 0; r < 8; ++r) {
            const int s   = q0 + r + hi8;
            const int col = h * D_HEAD + nt * 16 + lo;
            const float val = oacc[nt][r] / lsum[r];
            attn_out[((size_t)b * SEQ + s) * D_MODEL + col] = (__bf16)val;
        }
}

// ---------------------------------------------------------------------------
extern "C" void kernel_launch(void* const* d_in, const int* in_sizes, int n_in,
                              void* d_out, int out_size, void* d_ws, size_t ws_size,
                              hipStream_t stream) {
    const float* queries = (const float*)d_in[0];
    const float* keys    = (const float*)d_in[1];
    const float* values  = (const float*)d_in[2];
    const int*   mask    = (const int*)d_in[3];
    const float* Wq = (const float*)d_in[4];
    const float* bq = (const float*)d_in[5];
    const float* Wk = (const float*)d_in[6];
    const float* bk = (const float*)d_in[7];
    const float* Wv = (const float*)d_in[8];
    const float* bv = (const float*)d_in[9];
    const float* Wo = (const float*)d_in[10];
    const float* bo = (const float*)d_in[11];

    char* ws = (char*)d_ws;
    __bf16* wpack = (__bf16*)ws;                             // 4*512*512 bf16 = 2 MiB
    size_t off = (size_t)4 * 512 * 512 * sizeof(__bf16);
    const size_t act = (size_t)M_TOTAL * D_MODEL * sizeof(__bf16);  // 8 MiB each
    __bf16* qbf  = (__bf16*)(ws + off); off += act;
    __bf16* kbf  = (__bf16*)(ws + off); off += act;
    __bf16* vtbf = (__bf16*)(ws + off); off += act;
    __bf16* attn = (__bf16*)(ws + off); off += act;

    pack_weights<<<dim3(512, 4), 32, 0, stream>>>(Wq, Wk, Wv, Wo, wpack);

    dim3 ggrid(M_TOTAL / 64, D_MODEL / 64);
    gemm_wmma<float><<<ggrid, 128, 0, stream>>>(queries, wpack + (size_t)0 * 512 * 512, bq, qbf, 0);
    gemm_wmma<float><<<ggrid, 128, 0, stream>>>(keys,    wpack + (size_t)1 * 512 * 512, bk, kbf, 0);
    gemm_wmma<float><<<ggrid, 128, 0, stream>>>(values,  wpack + (size_t)2 * 512 * 512, bv, vtbf, 1);

    // 4096 q-tile waves (32 bh * 128 q-tiles), 4 waves per block
    flash_attn<<<1024, 128, 0, stream>>>(qbf, kbf, vtbf, mask, attn);

    gemm_wmma<__bf16><<<ggrid, 128, 0, stream>>>(attn, wpack + (size_t)3 * 512 * 512, bo, d_out, 2);
}
